// VisualFusion_76184129896913
// MI455X (gfx1250) — matrix operand order repaired and checked by reading, not compiled, not run on hardware
//
#include <hip/hip_runtime.h>

typedef __attribute__((ext_vector_type(16))) _Float16 v16h;
typedef __attribute__((ext_vector_type(8)))  float    v8f;

struct __align__(16) U128 { unsigned int v[4]; };
union HVec { U128 u[2]; v16h h; };

// ---- CDNA5 async global->LDS copy (ASYNCcnt-tracked DMA), per-lane 16B ----
__device__ __forceinline__ void async_copy_b128(unsigned ldsOff, const _Float16* g) {
    asm volatile("global_load_async_to_lds_b128 %0, %1, off"
                 :: "v"(ldsOff), "v"(g) : "memory");
}
__device__ __forceinline__ void async_waitcnt0() {
    asm volatile("s_wait_asynccnt 0x0" ::: "memory");
}
// LDS byte offset = low 32 bits of the generic (flat) shared-aperture address
__device__ __forceinline__ unsigned lds_off(const void* p) {
    return (unsigned)(size_t)p;
}

// ---------------------------------------------------------------------------
// Mixed-precision WMMA GEMM:  Y[b] = W[b] (MxK, row-major) * X[b] ([N][K]) ...
//   - 256 threads (8 waves), tile 64(M) x 64(N), K step 32
//   - wave tile 16(M) x 32(N): two v_wmma_f32_16x16x32_f16 per K step
//   - double-buffered LDS, staged with global_load_async_to_lds_b128
//   - optional bias / f32 residual add; f32 out and/or f16 out (opt transposed)
// ---------------------------------------------------------------------------
#define LSTR 40           // padded LDS row stride (halfs): 16B-aligned rows
#define LBUF (64 * LSTR)  // one tile buffer

__global__ __launch_bounds__(256) void k_gemm_f16(
    int M, int N, int K,
    const _Float16* __restrict__ W, long wStride,
    const _Float16* __restrict__ X, long xStride,     // X is [N][K] (transposed)
    const float* __restrict__ bias,
    const float* __restrict__ add, long addStride,
    float* __restrict__ Y, long yStride,
    _Float16* __restrict__ Y16, long y16Stride, int y16T, long y16Row)
{
    __shared__ __align__(16) _Float16 lA[2 * LBUF];
    __shared__ __align__(16) _Float16 lB[2 * LBUF];

    const int tid  = threadIdx.x;
    const int b    = blockIdx.z;
    const _Float16* Wb = W + (long)b * wStride;
    const _Float16* Xb = X + (long)b * xStride;
    const int m0 = blockIdx.y * 64;
    const int n0 = blockIdx.x * 64;

    const int wave = tid >> 5, lane = tid & 31;
    const int wm = wave >> 1, wn = wave & 1;

    // staging indices: 64 rows x 32 halfs, one 16B chunk per thread per matrix
    const int ar = tid >> 2, ac = (tid & 3) << 3;

    v8f acc[2] = {};

    auto stage = [&](int buf, int kt) {
        _Float16* sA = lA + buf * LBUF;
        _Float16* sB = lB + buf * LBUF;
        const int gm = m0 + ar;
        if (gm < M) {
            async_copy_b128(lds_off(sA + ar * LSTR + ac), Wb + (long)gm * K + kt + ac);
        } else {
            U128 z = {{0u, 0u, 0u, 0u}};
            *(U128*)(sA + ar * LSTR + ac) = z;          // zero-fill OOB rows (M=32 case)
        }
        async_copy_b128(lds_off(sB + ar * LSTR + ac), Xb + (long)(n0 + ar) * K + kt + ac);
    };

    stage(0, 0);
    int cur = 0;
    for (int kt = 0; kt < K; kt += 32) {
        async_waitcnt0();        // my async copies into buf `cur` landed
        __syncthreads();         // everyone's copies landed; prev-buf reads done
        if (kt + 32 < K) stage(cur ^ 1, kt + 32);   // overlap DMA with WMMA

        const _Float16* sA = lA + cur * LBUF;
        const _Float16* sB = lB + cur * LBUF;

        // A fragment (16x32): lanes 0-15 K{0..7,16..23}, lanes 16-31 K{8..15,24..31}
        const int arow = lane & 15;
        const int kOff = (lane >> 4) << 3;
        const _Float16* pA = sA + (wm * 16 + arow) * LSTR + kOff;
        HVec a;
        a.u[0] = *(const U128*)(pA);
        a.u[1] = *(const U128*)(pA + 16);

        // B fragments (32x16): lane col = lane&15; lanes<16 K{0..7,16..23}, lanes>=16 K{8..15,24..31}
        const int bcol  = lane & 15;
        const int bkOff = (lane >> 4) << 3;
        #pragma unroll
        for (int g = 0; g < 2; ++g) {
            const _Float16* pB = sB + (wn * 32 + g * 16 + bcol) * LSTR + bkOff;
            HVec bb;
            bb.u[0] = *(const U128*)(pB);
            bb.u[1] = *(const U128*)(pB + 16);
            acc[g] = __builtin_amdgcn_wmma_f32_16x16x32_f16(
                false, a.h, false, bb.h, (short)0, acc[g], false, false);
        }
        cur ^= 1;
    }

    // epilogue: C/D layout lane n = lane&15, M = (lane<16?0:8)+r
    const int ncol = lane & 15;
    const int mOff = (lane >> 4) << 3;
    #pragma unroll
    for (int g = 0; g < 2; ++g) {
        #pragma unroll
        for (int r = 0; r < 8; ++r) {
            const int m = m0 + wm * 16 + mOff + r;
            const int n = n0 + wn * 32 + g * 16 + ncol;
            if (m < M) {
                float v = acc[g][r];
                if (bias) v += bias[m];
                if (add)  v += add[(long)b * addStride + (long)m * N + n];
                if (Y)    Y[(long)b * yStride + (long)m * N + n] = v;
                if (Y16) {
                    if (y16T) Y16[(long)b * y16Stride + (long)n * y16Row + m] = (_Float16)v;
                    else      Y16[(long)b * y16Stride + (long)m * N + n] = (_Float16)v;
                }
            }
        }
    }
}

// ---------------------------------------------------------------------------
__global__ void k_f32_to_f16(const float* __restrict__ s, _Float16* __restrict__ d, int n) {
    int i = blockIdx.x * 256 + threadIdx.x;
    if (i < n) d[i] = (_Float16)s[i];
}

// transposing convert: dst[b][n][k] = (f16) src[b][k][n]
__global__ void k_f32_to_f16_t(const float* __restrict__ src, _Float16* __restrict__ dst,
                               int K, int N, int total)
{
    int idx = blockIdx.x * 256 + threadIdx.x;
    if (idx >= total) return;
    const int kn = K * N;
    const int b = idx / kn, rem = idx % kn;
    const int n = rem / K, k = rem % K;
    dst[idx] = (_Float16)src[(long)b * kn + (long)k * N + n];
}

// softmax over rows of length n; writes f16 probs
__global__ __launch_bounds__(256) void k_softmax_row(
    const float* __restrict__ S, _Float16* __restrict__ P, int n)
{
    __shared__ float red[256];
    const long row = blockIdx.x;
    const float* s = S + row * (long)n;
    _Float16*    p = P + row * (long)n;
    const int tid = threadIdx.x;

    float mx = -3.4e38f;
    for (int i = tid; i < n; i += 256) mx = fmaxf(mx, s[i]);
    red[tid] = mx; __syncthreads();
    for (int st = 128; st > 0; st >>= 1) { if (tid < st) red[tid] = fmaxf(red[tid], red[tid + st]); __syncthreads(); }
    mx = red[0]; __syncthreads();

    float sum = 0.f;
    for (int i = tid; i < n; i += 256) sum += __expf(s[i] - mx);
    red[tid] = sum; __syncthreads();
    for (int st = 128; st > 0; st >>= 1) { if (tid < st) red[tid] += red[tid + st]; __syncthreads(); }
    const float inv = 1.f / red[0];
    for (int i = tid; i < n; i += 256) p[i] = (_Float16)(__expf(s[i] - mx) * inv);
}

// involution: out[c][hw] = sum_t df[c*9+t][hw] * xpad[c][h+dh][w+dw]   (one batch)
__global__ void k_dynconv(const float* __restrict__ df, const float* __restrict__ x,
                          float* __restrict__ out)
{
    const int idx = blockIdx.x * 256 + threadIdx.x;     // 256*1024
    const int c = idx >> 10, hw = idx & 1023, h = hw >> 5, w = hw & 31;
    float acc = 0.f;
    #pragma unroll
    for (int t = 0; t < 9; ++t) {
        const int hh = h + t / 3 - 1, ww = w + t % 3 - 1;
        const float xv = (hh >= 0 && hh < 32 && ww >= 0 && ww < 32) ? x[c * 1024 + hh * 32 + ww] : 0.f;
        acc += df[(c * 9 + t) * 1024 + hw] * xv;
    }
    out[idx] = acc;
}

// per (b,c): avg & max over 1024 pixels
__global__ __launch_bounds__(256) void k_chanstats(const float* __restrict__ x,
                                                   float* __restrict__ avg, float* __restrict__ mx)
{
    __shared__ float rs[256], rm[256];
    const long bc = blockIdx.x;
    const float* p = x + bc * 1024;
    const int tid = threadIdx.x;
    float s = 0.f, m = -3.4e38f;
    for (int i = tid; i < 1024; i += 256) { const float v = p[i]; s += v; m = fmaxf(m, v); }
    rs[tid] = s; rm[tid] = m; __syncthreads();
    for (int st = 128; st > 0; st >>= 1) {
        if (tid < st) { rs[tid] += rs[tid + st]; rm[tid] = fmaxf(rm[tid], rm[tid + st]); }
        __syncthreads();
    }
    if (tid == 0) { avg[bc] = rs[0] * (1.f / 1024.f); mx[bc] = rm[0]; }
}

// SE: s[b][o] = sigmoid( w2 * (relu(w1*avg) + relu(w1*max)) )
__global__ __launch_bounds__(256) void k_se(const float* __restrict__ avg, const float* __restrict__ mx,
                                            const float* __restrict__ w1, const float* __restrict__ w2,
                                            float* __restrict__ s)
{
    __shared__ float h1[16];
    const int b = blockIdx.x, tid = threadIdx.x;
    if (tid < 16) {
        float aa = 0.f, mm = 0.f;
        for (int c = 0; c < 256; ++c) {
            const float w = w1[tid * 256 + c];
            aa += w * avg[b * 256 + c];
            mm += w * mx[b * 256 + c];
        }
        h1[tid] = fmaxf(aa, 0.f) + fmaxf(mm, 0.f);
    }
    __syncthreads();
    float acc = 0.f;
    #pragma unroll
    for (int r = 0; r < 16; ++r) acc += w2[tid * 16 + r] * h1[r];
    s[b * 256 + tid] = 1.f / (1.f + __expf(-acc));
}

// per (b,hw): mean & max over channels of dyn*sSE
__global__ void k_spatstats(const float* __restrict__ x, const float* __restrict__ s,
                            float* __restrict__ mean, float* __restrict__ mx)
{
    const int idx = blockIdx.x * 256 + threadIdx.x;  // 8*1024
    const int b = idx >> 10, hw = idx & 1023;
    float sum = 0.f, m = -3.4e38f;
    for (int c = 0; c < 256; ++c) {
        const float v = x[((long)b * 256 + c) * 1024 + hw] * s[b * 256 + c];
        sum += v; m = fmaxf(m, v);
    }
    mean[idx] = sum * (1.f / 256.f); mx[idx] = m;
}

// 7x7 conv (2->1, pad 3) + sigmoid
__global__ void k_sconv7(const float* __restrict__ mean, const float* __restrict__ mx,
                         const float* __restrict__ wsa, float* __restrict__ sg)
{
    const int idx = blockIdx.x * 256 + threadIdx.x;  // 8*1024
    const int b = idx >> 10, hw = idx & 1023, h = hw >> 5, w = hw & 31;
    float acc = 0.f;
    for (int i = 0; i < 7; ++i) {
        const int hh = h + i - 3;
        if (hh < 0 || hh >= 32) continue;
        for (int j = 0; j < 7; ++j) {
            const int ww = w + j - 3;
            if (ww < 0 || ww >= 32) continue;
            const int m = b * 1024 + hh * 32 + ww;
            acc += mean[m] * wsa[i * 7 + j] + mx[m] * wsa[49 + i * 7 + j];
        }
    }
    sg[idx] = 1.f / (1.f + __expf(-acc));
}

// rec = dyn * sSE[b][c] * sg[b][hw]
__global__ void k_rec(const float* __restrict__ x, const float* __restrict__ s,
                      const float* __restrict__ sg, float* __restrict__ out)
{
    const int idx = blockIdx.x * 256 + threadIdx.x;  // 8*256*1024
    const int b = idx >> 18, c = (idx >> 10) & 255, hw = idx & 1023;
    out[idx] = x[idx] * s[b * 256 + c] * sg[b * 1024 + hw];
}

// transposed im2col (3x3, pad 1) -> f16 [1024][2304]   (one batch)
__global__ void k_im2col_t(const float* __restrict__ rec, _Float16* __restrict__ col)
{
    const int idx = blockIdx.x * 256 + threadIdx.x;  // 1024*2304
    const int hw = idx / 2304, r = idx % 2304;
    const int c = r / 9, t = r % 9, h = hw >> 5, w = hw & 31;
    const int hh = h + t / 3 - 1, ww = w + t % 3 - 1;
    const float v = (hh >= 0 && hh < 32 && ww >= 0 && ww < 32) ? rec[c * 1024 + hh * 32 + ww] : 0.f;
    col[idx] = (_Float16)v;
}

// gate + blend + channel LayerNorm; one block per pixel, thread per channel
__global__ __launch_bounds__(256) void k_final(const float* __restrict__ glin,
                                               const float* __restrict__ ra, const float* __restrict__ rb,
                                               const float* __restrict__ lnw, const float* __restrict__ lnb,
                                               float* __restrict__ out)
{
    __shared__ float rs[256], rq[256];
    const int bp = blockIdx.x;             // b*1024 + p
    const int b = bp >> 10, p = bp & 1023;
    const int c = threadIdx.x;
    const long idx = ((long)b * 256 + c) * 1024 + p;
    const float g   = 1.f / (1.f + __expf(-glin[idx]));
    const float val = g * ra[idx] + (1.f - g) * rb[idx];
    rs[c] = val; rq[c] = val * val; __syncthreads();
    for (int st = 128; st > 0; st >>= 1) {
        if (c < st) { rs[c] += rs[c + st]; rq[c] += rq[c + st]; }
        __syncthreads();
    }
    const float mu  = rs[0] * (1.f / 256.f);
    const float var = rq[0] * (1.f / 256.f) - mu * mu;
    out[idx] = (val - mu) * rsqrtf(var + 1e-5f) * lnw[c] + lnb[c];
}

// ---------------------------------------------------------------------------
static inline void launch_gemm(hipStream_t st, int M, int N, int K,
                               const _Float16* W, long ws, const _Float16* X, long xs,
                               const float* bias, const float* add, long as,
                               float* Y, long ys,
                               _Float16* Y16, long y16s, int y16T, long y16Row, int batches)
{
    dim3 grid(N / 64, (M + 63) / 64, batches);
    k_gemm_f16<<<grid, 256, 0, st>>>(M, N, K, W, ws, X, xs, bias, add, as, Y, ys, Y16, y16s, y16T, y16Row);
}

extern "C" void kernel_launch(void* const* d_in, const int* in_sizes, int n_in,
                              void* d_out, int out_size, void* d_ws, size_t ws_size,
                              hipStream_t stream)
{
    (void)in_sizes; (void)n_in; (void)out_size; (void)ws_size;
    const long NPIX = 1024, BATCH = 8, OD = 256, RD = 512, DDIM = 256;
    const long PB = OD * NPIX;          // 262144 per-batch plane
    const long NN = NPIX * NPIX;        // 1048576

    const float* b_res   = (const float*)d_in[5];
    const float* b_dino  = (const float*)d_in[7];
    const float* b_q     = (const float*)d_in[9];
    const float* b_k     = (const float*)d_in[11];
    const float* b_v     = (const float*)d_in[13];
    const float* b_gen   = (const float*)d_in[15];
    const float* w_ca1   = (const float*)d_in[16];
    const float* w_ca2   = (const float*)d_in[17];
    const float* w_sa    = (const float*)d_in[18];
    const float* b_resid = (const float*)d_in[20];
    const float* b_gate  = (const float*)d_in[22];
    const float* ln_w    = (const float*)d_in[23];
    const float* ln_b    = (const float*)d_in[24];

    char* ws = (char*)d_ws;
    size_t off = 0;
    auto alloc = [&](size_t bytes) -> char* {
        char* p = ws + off; off += (bytes + 255) & ~(size_t)255; return p;
    };

    // persistent
    _Float16* wres16    = (_Float16*)alloc(OD * RD * 2);
    _Float16* wdino16   = (_Float16*)alloc(OD * DDIM * 2);
    _Float16* wq16      = (_Float16*)alloc(32 * OD * 2);
    _Float16* wk16      = (_Float16*)alloc(32 * OD * 2);
    _Float16* wv16      = (_Float16*)alloc(OD * OD * 2);
    _Float16* wgen16    = (_Float16*)alloc(2304 * OD * 2);
    _Float16* wresid16  = (_Float16*)alloc(OD * 2304 * 2);
    _Float16* wgate16   = (_Float16*)alloc(OD * 512 * 2);
    float*    resA32    = (float*)alloc(BATCH * PB * 4);
    float*    resB32    = (float*)alloc(BATCH * PB * 4);
    _Float16* concatT16 = (_Float16*)alloc(BATCH * NPIX * 512 * 2);  // [b][n][512]
    float*    g32       = (float*)alloc(BATCH * PB * 4);

    // per-pass scratch (reused A then B); all f16 activations stored [N][K]
    _Float16* featT16  = (_Float16*)alloc(BATCH * NPIX * RD * 2);
    _Float16* dinoT16  = (_Float16*)alloc(BATCH * NPIX * DDIM * 2);
    float*    fA32     = (float*)alloc(BATCH * PB * 4);
    _Float16* fAT16    = (_Float16*)alloc(BATCH * NPIX * OD * 2);
    _Float16* dAT16    = (_Float16*)alloc(BATCH * NPIX * OD * 2);
    _Float16* qT16     = (_Float16*)alloc(BATCH * NPIX * 32 * 2);
    _Float16* kT16     = (_Float16*)alloc(BATCH * NPIX * 32 * 2);
    _Float16* v16      = (_Float16*)alloc(BATCH * PB * 2);           // [c][n] (W operand)
    float*    S32      = (float*)alloc(BATCH * NN * 4);
    _Float16* attn16   = (_Float16*)alloc(BATCH * NN * 2);           // [m][n] == [N][K]
    float*    fused32  = (float*)alloc(BATCH * PB * 4);
    _Float16* fusedT16 = (_Float16*)alloc(BATCH * NPIX * OD * 2);
    float*    df32     = (float*)alloc(2304 * NPIX * 4);             // one batch at a time
    float*    dyn32    = (float*)alloc(BATCH * PB * 4);
    float*    cavg     = (float*)alloc(BATCH * OD * 4);
    float*    cmax     = (float*)alloc(BATCH * OD * 4);
    float*    sSE      = (float*)alloc(BATCH * OD * 4);
    float*    smean    = (float*)alloc(BATCH * NPIX * 4);
    float*    smax     = (float*)alloc(BATCH * NPIX * 4);
    float*    sg       = (float*)alloc(BATCH * NPIX * 4);
    float*    rec32    = (float*)alloc(BATCH * PB * 4);
    _Float16* colT16   = (_Float16*)alloc(NPIX * 2304 * 2);          // one batch at a time

    auto cvt = [&](const void* src, _Float16* dst, long n) {
        k_f32_to_f16<<<dim3((unsigned)((n + 255) / 256)), 256, 0, stream>>>((const float*)src, dst, (int)n);
    };

    // weights -> f16 (w_resid OIHW flattens to [256][2304] directly)
    cvt(d_in[4],  wres16,   OD * RD);
    cvt(d_in[6],  wdino16,  OD * DDIM);
    cvt(d_in[8],  wq16,     32 * OD);
    cvt(d_in[10], wk16,     32 * OD);
    cvt(d_in[12], wv16,     OD * OD);
    cvt(d_in[14], wgen16,   2304 * OD);
    cvt(d_in[19], wresid16, OD * 2304);
    cvt(d_in[21], wgate16,  OD * 512);

    for (int p = 0; p < 2; ++p) {
        const float* feat = (const float*)d_in[p == 0 ? 0 : 1];
        const float* dino = (const float*)d_in[p == 0 ? 2 : 3];
        float* resOut = (p == 0) ? resA32 : resB32;

        // inputs -> f16 [N][K]
        k_f32_to_f16_t<<<dim3(16384), 256, 0, stream>>>(feat, featT16, (int)RD, (int)NPIX,
                                                        (int)(BATCH * RD * NPIX));
        k_f32_to_f16_t<<<dim3(8192), 256, 0, stream>>>(dino, dinoT16, (int)DDIM, (int)NPIX,
                                                       (int)(BATCH * DDIM * NPIX));

        // projections (f16 copies written transposed for downstream GEMMs)
        launch_gemm(stream, 256, 1024, 512, wres16, 0, featT16, NPIX * RD,
                    b_res, nullptr, 0, fA32, PB, fAT16, NPIX * OD, 1, OD, 8);
        launch_gemm(stream, 256, 1024, 256, wdino16, 0, dinoT16, NPIX * DDIM,
                    b_dino, nullptr, 0, nullptr, 0, dAT16, NPIX * OD, 1, OD, 8);

        // q, k (both transposed [n][32]), v ([c][n], W operand of O-GEMM)
        launch_gemm(stream, 32, 1024, 256, wq16, 0, fAT16, NPIX * OD,
                    b_q, nullptr, 0, nullptr, 0, qT16, NPIX * 32, 1, 32, 8);
        launch_gemm(stream, 32, 1024, 256, wk16, 0, dAT16, NPIX * OD,
                    b_k, nullptr, 0, nullptr, 0, kT16, NPIX * 32, 1, 32, 8);
        launch_gemm(stream, 256, 1024, 256, wv16, 0, dAT16, NPIX * OD,
                    b_v, nullptr, 0, nullptr, 0, v16, PB, 0, 0, 8);

        // S = qT * k  (per-batch "weight" = qT [1024][32], X = kT [1024][32])
        launch_gemm(stream, 1024, 1024, 32, qT16, NPIX * 32, kT16, NPIX * 32,
                    nullptr, nullptr, 0, S32, NN, nullptr, 0, 0, 0, 8);
        k_softmax_row<<<dim3(8 * 1024), 256, 0, stream>>>(S32, attn16, 1024);

        // O = V * attn^T + feat   (attn16 [m][n] is exactly the [N][K] layout)
        launch_gemm(stream, 256, 1024, 1024, v16, PB, attn16, NN,
                    nullptr, fA32, PB, fused32, PB, fusedT16, NPIX * OD, 1, OD, 8);

        // dynamic filters + involution (per batch to bound df size)
        for (int b = 0; b < 8; ++b) {
            launch_gemm(stream, 2304, 1024, 256, wgen16, 0, fusedT16 + b * NPIX * OD, 0,
                        b_gen, nullptr, 0, df32, 0, nullptr, 0, 0, 0, 1);
            k_dynconv<<<dim3(1024), 256, 0, stream>>>(df32, fused32 + b * PB, dyn32 + b * PB);
        }

        // channel attention + spatial attention
        k_chanstats<<<dim3(8 * 256), 256, 0, stream>>>(dyn32, cavg, cmax);
        k_se<<<dim3(8), 256, 0, stream>>>(cavg, cmax, w_ca1, w_ca2, sSE);
        k_spatstats<<<dim3(32), 256, 0, stream>>>(dyn32, sSE, smean, smax);
        k_sconv7<<<dim3(32), 256, 0, stream>>>(smean, smax, w_sa, sg);
        k_rec<<<dim3(8192), 256, 0, stream>>>(dyn32, sSE, sg, rec32);

        // residual 3x3 conv as implicit GEMM (per batch); f16 scattered into concat rows
        for (int b = 0; b < 8; ++b) {
            k_im2col_t<<<dim3(9216), 256, 0, stream>>>(rec32 + b * PB, colT16);
            launch_gemm(stream, 256, 1024, 2304, wresid16, 0, colT16, 0,
                        b_resid, nullptr, 0, resOut + b * PB, 0,
                        concatT16 + b * NPIX * 512 + p * 256, 0, 1, 512, 1);
        }
    }

    // gate = w_gate * concat(resA,resB)  (sigmoid applied in final kernel)
    launch_gemm(stream, 256, 1024, 512, wgate16, 0, concatT16, NPIX * 512,
                b_gate, nullptr, 0, g32, PB, nullptr, 0, 0, 0, 8);

    // blend + LayerNorm
    k_final<<<dim3(8192), 256, 0, stream>>>(g32, resA32, resB32, ln_w, ln_b, (float*)d_out);
}
